// StructurePredictor_25469156065592
// MI455X (gfx1250) — matrix operand order repaired and checked by reading, not compiled, not run on hardware
//
#include <hip/hip_runtime.h>
#include <hip/hip_bf16.h>
#include <math.h>

// Problem dims (match reference)
#define B_   8
#define S_   512
#define H_   768
#define L_   12
#define NH_  8
#define DH_  96
#define FF_  3072
#define MTOT (B_ * S_)   // 4096 rows

typedef __attribute__((ext_vector_type(16))) _Float16 v16h;
typedef __attribute__((ext_vector_type(8)))  _Float16 v8h;
typedef __attribute__((ext_vector_type(8)))  float    v8f;

// GEMM tiling: 8 waves (256 thr); tile 128x128x32; wave = 32x64 patch = 2x4 WMMA
#define BM   128
#define BN   128
#define BK   32
#define LDST 40   // LDS row stride in halves (80B: 16B aligned, hits all 64 banks)

// ---------------------------------------------------------------------------
// Generic batched WMMA GEMM: C = alpha * (A @ B) + bias  [+ ReLU]
//   A: fp32 [M,K] row-major (lda), staged fp32->f16 into LDS
//   B: TRANSB==0 -> fp32 [K,N] (ldb);  TRANSB==1 -> fp32 [N,K] (ldb)
//   batching: z = zo*nInner + zi, pointer offsets (sAo,sAi)/(sBo,sBi)/(sCo,sCi)
//   requires K % 32 == 0 (true for all uses: 768, 3072, 512, 96)
// ---------------------------------------------------------------------------
template <int TRANSB>
__global__ void __launch_bounds__(256)
gemm_wmma_f16(const float* __restrict__ Ag, const float* __restrict__ Bg,
              const float* __restrict__ bias, float* __restrict__ Cg,
              int M, int N, int K, int lda, int ldb, int ldc,
              long long sAo, long long sAi, long long sBo, long long sBi,
              long long sCo, long long sCi, int nInner,
              float alpha, int relu)
{
    __shared__ _Float16 lA[BM * LDST];
    __shared__ _Float16 lB[BN * LDST];   // stored n-major: lB[n*LDST + k]

    const int z  = blockIdx.z;
    const int zo = z / nInner;
    const int zi = z - zo * nInner;
    const float* A  = Ag + zo * sAo + zi * sAi;
    const float* Bm = Bg + zo * sBo + zi * sBi;
    float*       C  = Cg + zo * sCo + zi * sCi;

    const int m0   = blockIdx.y * BM;
    const int n0   = blockIdx.x * BN;
    const int tid  = threadIdx.x;
    const int lane = tid & 31;
    const int wave = tid >> 5;
    const int wr   = (wave >> 1) * 32;   // wave row offset in tile (0,32,64,96)
    const int wc   = (wave & 1) * 64;    // wave col offset in tile (0,64)

    v8f acc[2][4];
    #pragma unroll
    for (int i = 0; i < 2; ++i)
        #pragma unroll
        for (int j = 0; j < 4; ++j)
            #pragma unroll
            for (int e = 0; e < 8; ++e) acc[i][j][e] = 0.f;

    const int ml    = lane & 15;
    const int kaOff = (lane >> 4) << 3;   // A-frag K base (0 / 8)  per ISA 16-bit A 16x32
    const int kbOff = (lane >> 4) << 3;   // B-frag K base (0 / 8) per ISA 16-bit B 32x16

    // staging indices (256 threads; 16 fp32 values per thread per tile)
    const int ar  = tid >> 1;           // A row 0..127
    const int aks = (tid & 1) << 4;     // A k-segment 0/16
    const int gmA = m0 + ar;
    const int br  = tid >> 1;           // B row (TRANSB==1)
    const int bks = (tid & 1) << 4;
    const int bn  = tid & 127;          // B col (TRANSB==0)
    const int bkb = (tid >> 7) << 4;

    auto fetchA = [&](int kk, float* r) {
        if (gmA < M) {
            const float* src = A + (long long)gmA * lda + kk + aks;
            #pragma unroll
            for (int i = 0; i < 16; ++i) r[i] = src[i];
        } else {
            #pragma unroll
            for (int i = 0; i < 16; ++i) r[i] = 0.f;
        }
    };
    auto fetchB = [&](int kk, float* r) {
        if (TRANSB) {
            const int gn = n0 + br;
            if (gn < N) {
                const float* src = Bm + (long long)gn * ldb + kk + bks;
                #pragma unroll
                for (int i = 0; i < 16; ++i) r[i] = src[i];
            } else {
                #pragma unroll
                for (int i = 0; i < 16; ++i) r[i] = 0.f;
            }
        } else {
            const int gn = n0 + bn;
            #pragma unroll
            for (int i = 0; i < 16; ++i)
                r[i] = (gn < N) ? Bm[(long long)(kk + bkb + i) * ldb + gn] : 0.f;
        }
    };
    auto storeA = [&](const float* r) {
        _Float16* dst = &lA[ar * LDST + aks];
        #pragma unroll
        for (int i = 0; i < 16; ++i) dst[i] = (_Float16)r[i];
    };
    auto storeB = [&](const float* r) {
        _Float16* dst = TRANSB ? &lB[br * LDST + bks] : &lB[bn * LDST + bkb];
        #pragma unroll
        for (int i = 0; i < 16; ++i) dst[i] = (_Float16)r[i];
    };

    // ---- prologue: stage tile 0 ----
    float ra[16], rb[16];
    fetchA(0, ra); fetchB(0, rb);
    storeA(ra);    storeB(rb);
    __syncthreads();

    for (int k0 = 0; k0 < K; k0 += BK) {
        const bool nxt = (k0 + BK) < K;
        // issue next tile's global loads first -> overlap with WMMA burst
        if (nxt) { fetchA(k0 + BK, ra); fetchB(k0 + BK, rb); }

        // ---- fragments from LDS, 2x4 WMMA ----
        v16h af[2], bf[4];
        #pragma unroll
        for (int mi = 0; mi < 2; ++mi) {
            const _Float16* pa = &lA[(wr + mi * 16 + ml) * LDST + kaOff];
            v8h lo = *(const v8h*)pa;          // K = kaOff .. kaOff+7
            v8h hi = *(const v8h*)(pa + 16);   // K = 16+kaOff .. 16+kaOff+7
            v16h a;
            #pragma unroll
            for (int i = 0; i < 8; ++i) { a[i] = lo[i]; a[i + 8] = hi[i]; }
            af[mi] = a;
        }
        #pragma unroll
        for (int ni = 0; ni < 4; ++ni) {
            const _Float16* pb = &lB[(wc + ni * 16 + ml) * LDST + kbOff];
            v8h lo = *(const v8h*)pb;          // K = kbOff .. kbOff+7
            v8h hi = *(const v8h*)(pb + 16);   // K = 16+kbOff .. 16+kbOff+7
            v16h b;
            #pragma unroll
            for (int i = 0; i < 8; ++i) { b[i] = lo[i]; b[i + 8] = hi[i]; }
            bf[ni] = b;
        }
        #pragma unroll
        for (int mi = 0; mi < 2; ++mi)
            #pragma unroll
            for (int ni = 0; ni < 4; ++ni)
                acc[mi][ni] = __builtin_amdgcn_wmma_f32_16x16x32_f16(
                    false, af[mi], false, bf[ni], (short)0, acc[mi][ni], false, false);

        __syncthreads();
        if (nxt) { storeA(ra); storeB(rb); }
        __syncthreads();
    }

    // ---- epilogue: f32 C/D layout (VGPR r -> m = r + 8*(lane>=16)) ----
    const int rg = (lane >> 4) << 3;
    #pragma unroll
    for (int mi = 0; mi < 2; ++mi) {
        #pragma unroll
        for (int ni = 0; ni < 4; ++ni) {
            const int n = n0 + wc + ni * 16 + ml;
            if (n >= N) continue;
            const float bv = bias ? bias[n] : 0.f;
            #pragma unroll
            for (int r = 0; r < 8; ++r) {
                const int m = m0 + wr + mi * 16 + rg + r;
                if (m >= M) continue;
                float v = alpha * acc[mi][ni][r] + bv;
                if (relu) v = fmaxf(v, 0.f);
                C[(long long)m * ldc + n] = v;
            }
        }
    }
}

// ---------------------------------------------------------------------------
// Encoder: hidden = relu(features @ w_enc + b_enc), K = 20 (too small for WMMA)
// ---------------------------------------------------------------------------
__global__ void encoder_kernel(const float* __restrict__ feat,
                               const float* __restrict__ w,
                               const float* __restrict__ b,
                               float* __restrict__ out, int M, int Kk, int Nn)
{
    long long idx = (long long)blockIdx.x * blockDim.x + threadIdx.x;
    if (idx >= (long long)M * Nn) return;
    int m = (int)(idx / Nn), n = (int)(idx % Nn);
    float acc = b[n];
    const float* fr = feat + (long long)m * Kk;
    for (int k = 0; k < Kk; ++k) acc = fmaf(fr[k], w[(long long)k * Nn + n], acc);
    out[idx] = fmaxf(acc, 0.f);
}

// ---------------------------------------------------------------------------
// Row softmax over length n (one wave32 per row, shuffle reductions)
// ---------------------------------------------------------------------------
__global__ void softmax_kernel(float* __restrict__ S, int n)
{
    float* p = S + (long long)blockIdx.x * n;
    const int lane = threadIdx.x;
    float mx = -3.0e38f;
    for (int i = lane; i < n; i += 32) mx = fmaxf(mx, p[i]);
    #pragma unroll
    for (int o = 16; o > 0; o >>= 1) mx = fmaxf(mx, __shfl_xor(mx, o));
    float sum = 0.f;
    for (int i = lane; i < n; i += 32) { float e = __expf(p[i] - mx); p[i] = e; sum += e; }
    #pragma unroll
    for (int o = 16; o > 0; o >>= 1) sum += __shfl_xor(sum, o);
    const float inv = 1.f / sum;
    for (int i = lane; i < n; i += 32) p[i] *= inv;
}

// ---------------------------------------------------------------------------
// hidden = LayerNorm(hidden + delta) * g + b  — one 256-thread block per row
// ---------------------------------------------------------------------------
__global__ void __launch_bounds__(256)
add_ln_kernel(float* __restrict__ hidden, const float* __restrict__ delta,
              const float* __restrict__ g, const float* __restrict__ b)
{
    const long long row = blockIdx.x;
    float* hp = hidden + row * H_;
    const float* dp = delta + row * H_;
    const int tid = threadIdx.x;
    __shared__ float red[256];

    float v[3]; float s = 0.f;
    #pragma unroll
    for (int i = 0; i < 3; ++i) { int c = tid + i * 256; v[i] = hp[c] + dp[c]; s += v[i]; }
    red[tid] = s; __syncthreads();
    for (int o = 128; o > 0; o >>= 1) { if (tid < o) red[tid] += red[tid + o]; __syncthreads(); }
    const float mu = red[0] * (1.f / H_);
    __syncthreads();
    float s2 = 0.f;
    #pragma unroll
    for (int i = 0; i < 3; ++i) { float d = v[i] - mu; s2 += d * d; }
    red[tid] = s2; __syncthreads();
    for (int o = 128; o > 0; o >>= 1) { if (tid < o) red[tid] += red[tid + o]; __syncthreads(); }
    const float rinv = rsqrtf(red[0] * (1.f / H_) + 1e-5f);
    #pragma unroll
    for (int i = 0; i < 3; ++i) { int c = tid + i * 256; hp[c] = (v[i] - mu) * rinv * g[c] + b[c]; }
}

// ---------------------------------------------------------------------------
// Tiny head GEMM (N=3): angles = a1 @ wa2 + ba2
// ---------------------------------------------------------------------------
__global__ void small_gemm_kernel(const float* __restrict__ A, const float* __restrict__ W,
                                  const float* __restrict__ b, float* __restrict__ C,
                                  int M, int Kk, int Nn)
{
    long long idx = (long long)blockIdx.x * blockDim.x + threadIdx.x;
    if (idx >= (long long)M * Nn) return;
    int m = (int)(idx / Nn), n = (int)(idx % Nn);
    float acc = b[n];
    const float* ar = A + (long long)m * Kk;
    for (int k = 0; k < Kk; ++k) acc = fmaf(ar[k], W[(long long)k * Nn + n], acc);
    C[idx] = acc;
}

// ---------------------------------------------------------------------------
// Sequential NeRF-style coordinate build (one thread per batch element)
// ---------------------------------------------------------------------------
__global__ void coords_kernel(const float* __restrict__ dist,
                              const float* __restrict__ ang,
                              float* __restrict__ out)
{
    const int b = threadIdx.x;
    if (b >= B_) return;
    const float* D = dist + (long long)b * S_ * H_;
    const float* A = ang  + (long long)b * S_ * 3;
    float* O = out + (long long)b * S_ * 3;

    const float rad120 = 2.0943951023931953f;
    float p3x = 0.f, p3y = 0.f, p3z = 0.f;                       // c0
    float p2x = 3.8f, p2y = 0.f, p2z = 0.f;                      // c1
    float p1x = 3.8f + 3.8f * cosf(rad120);                      // c2
    float p1y = 3.8f * sinf(rad120);
    float p1z = 0.f;
    O[0] = p3x; O[1] = p3y; O[2] = p3z;
    O[3] = p2x; O[4] = p2y; O[5] = p2z;
    O[6] = p1x; O[7] = p1y; O[8] = p1z;

    for (int i = 3; i < S_; ++i) {
        const float r  = D[(long long)i * H_ + (i - 1)];
        const float th = A[i * 3 + 1];
        const float ch = A[i * 3 + 2];
        float e1x = p1x - p2x, e1y = p1y - p2y, e1z = p1z - p2z;
        float inv = rsqrtf(e1x * e1x + e1y * e1y + e1z * e1z);
        e1x *= inv; e1y *= inv; e1z *= inv;
        float ux = p3x - p2x, uy = p3y - p2y, uz = p3z - p2z;
        inv = rsqrtf(ux * ux + uy * uy + uz * uz);
        ux *= inv; uy *= inv; uz *= inv;
        float e3x = e1y * uz - e1z * uy;
        float e3y = e1z * ux - e1x * uz;
        float e3z = e1x * uy - e1y * ux;
        inv = rsqrtf(e3x * e3x + e3y * e3y + e3z * e3z);
        e3x *= inv; e3y *= inv; e3z *= inv;
        float e2x = e3y * e1z - e3z * e1y;
        float e2y = e3z * e1x - e3x * e1z;
        float e2z = e3x * e1y - e3y * e1x;

        const float a1 = r * cosf(th);
        const float a2 = r * sinf(th) * cosf(ch);
        const float a3 = r * sinf(th) * sinf(ch);
        const float px = p1x + a1 * e1x + a2 * e2x + a3 * e3x;
        const float py = p1y + a1 * e1y + a2 * e2y + a3 * e3y;
        const float pz = p1z + a1 * e1z + a2 * e2z + a3 * e3z;
        O[i * 3 + 0] = px; O[i * 3 + 1] = py; O[i * 3 + 2] = pz;
        p3x = p2x; p3y = p2y; p3z = p2z;
        p2x = p1x; p2y = p1y; p2z = p1z;
        p1x = px;  p1y = py;  p1z = pz;
    }
}

// ---------------------------------------------------------------------------
// Host-side launch helper
// ---------------------------------------------------------------------------
static inline void launch_gemm(int transb, const float* A, const float* Bm, const float* bias,
                               float* C, int M, int N, int K, int lda, int ldb, int ldc,
                               long long sAo, long long sAi, long long sBo, long long sBi,
                               long long sCo, long long sCi, int nOuter, int nInner,
                               float alpha, int relu, hipStream_t stream)
{
    dim3 grid((N + BN - 1) / BN, (M + BM - 1) / BM, nOuter * nInner);
    if (transb)
        gemm_wmma_f16<1><<<grid, 256, 0, stream>>>(A, Bm, bias, C, M, N, K, lda, ldb, ldc,
                                                   sAo, sAi, sBo, sBi, sCo, sCi, nInner, alpha, relu);
    else
        gemm_wmma_f16<0><<<grid, 256, 0, stream>>>(A, Bm, bias, C, M, N, K, lda, ldb, ldc,
                                                   sAo, sAi, sBo, sBi, sCo, sCi, nInner, alpha, relu);
}

extern "C" void kernel_launch(void* const* d_in, const int* in_sizes, int n_in,
                              void* d_out, int out_size, void* d_ws, size_t ws_size,
                              hipStream_t stream)
{
    (void)in_sizes; (void)n_in; (void)out_size; (void)ws_size;

    const float* features = (const float*)d_in[0];
    const float* w_enc = (const float*)d_in[1];
    const float* b_enc = (const float*)d_in[2];
    const float* wqkv  = (const float*)d_in[3];
    const float* bqkv  = (const float*)d_in[4];
    const float* wo    = (const float*)d_in[5];
    const float* bo    = (const float*)d_in[6];
    const float* ln1_g = (const float*)d_in[7];
    const float* ln1_b = (const float*)d_in[8];
    const float* w1    = (const float*)d_in[9];
    const float* b1    = (const float*)d_in[10];
    const float* w2    = (const float*)d_in[11];
    const float* b2    = (const float*)d_in[12];
    const float* ln2_g = (const float*)d_in[13];
    const float* ln2_b = (const float*)d_in[14];
    const float* wd1   = (const float*)d_in[15];
    const float* bd1   = (const float*)d_in[16];
    const float* wd2   = (const float*)d_in[17];
    const float* bd2   = (const float*)d_in[18];
    const float* wa1   = (const float*)d_in[19];
    const float* ba1   = (const float*)d_in[20];
    const float* wa2   = (const float*)d_in[21];
    const float* ba2   = (const float*)d_in[22];

    // workspace carve-up (floats)
    float* ws = (float*)d_ws;
    size_t o = 0;
    float* hidden = ws + o;  o += (size_t)MTOT * H_;           // 3.1M
    float* qkv    = ws + o;  o += (size_t)MTOT * 3 * H_;       // 9.4M
    float* scores = ws + o;  o += (size_t)B_ * NH_ * S_ * S_;  // 16.8M
    float* ctx    = ws + o;  o += (size_t)MTOT * H_;           // 3.1M
    float* tmp1   = ws + o;  o += (size_t)MTOT * FF_;          // 12.6M (FF / head hidden)
    float* tmp2   = ws + o;                                    // 3.1M  (attn_out / ff_out)

    float* out_dist   = (float*)d_out;
    float* out_ang    = out_dist  + (size_t)MTOT * H_;
    float* out_coord  = out_ang   + (size_t)MTOT * 3;
    float* out_hidden = out_coord + (size_t)MTOT * 3;

    // ---- encoder ----
    {
        long long tot = (long long)MTOT * H_;
        encoder_kernel<<<(unsigned)((tot + 255) / 256), 256, 0, stream>>>(
            features, w_enc, b_enc, hidden, MTOT, 20, H_);
    }

    const float scale = 1.0f / sqrtf((float)DH_);

    for (int l = 0; l < L_; ++l) {
        // qkv = hidden @ wqkv[l] + bqkv[l]            [4096, 2304]
        launch_gemm(0, hidden, wqkv + (size_t)l * H_ * 3 * H_, bqkv + (size_t)l * 3 * H_, qkv,
                    MTOT, 3 * H_, H_, H_, 3 * H_, 3 * H_,
                    0, 0, 0, 0, 0, 0, 1, 1, 1.f, 0, stream);

        // scores[b,h] = (Q @ K^T) * scale   (batched over B*NH, B operand transposed)
        launch_gemm(1, qkv, qkv + H_, nullptr, scores,
                    S_, S_, DH_, 3 * H_, 3 * H_, S_,
                    (long long)S_ * 3 * H_, DH_,
                    (long long)S_ * 3 * H_, DH_,
                    (long long)NH_ * S_ * S_, (long long)S_ * S_,
                    B_, NH_, scale, 0, stream);

        // softmax over last dim
        softmax_kernel<<<B_ * NH_ * S_, 32, 0, stream>>>(scores, S_);

        // ctx[b,h] = attn @ V  -> scattered into [B,S,H] layout
        launch_gemm(0, scores, qkv + 2 * H_, nullptr, ctx,
                    S_, DH_, S_, S_, 3 * H_, H_,
                    (long long)NH_ * S_ * S_, (long long)S_ * S_,
                    (long long)S_ * 3 * H_, DH_,
                    (long long)S_ * H_, DH_,
                    B_, NH_, 1.f, 0, stream);

        // attn_out = ctx @ wo[l] + bo[l]
        launch_gemm(0, ctx, wo + (size_t)l * H_ * H_, bo + (size_t)l * H_, tmp2,
                    MTOT, H_, H_, H_, H_, H_,
                    0, 0, 0, 0, 0, 0, 1, 1, 1.f, 0, stream);

        // hidden = LN1(hidden + attn_out)
        add_ln_kernel<<<MTOT, 256, 0, stream>>>(hidden, tmp2, ln1_g + (size_t)l * H_, ln1_b + (size_t)l * H_);

        // ff = relu(hidden @ w1 + b1) @ w2 + b2
        launch_gemm(0, hidden, w1 + (size_t)l * H_ * FF_, b1 + (size_t)l * FF_, tmp1,
                    MTOT, FF_, H_, H_, FF_, FF_,
                    0, 0, 0, 0, 0, 0, 1, 1, 1.f, 1, stream);
        launch_gemm(0, tmp1, w2 + (size_t)l * FF_ * H_, b2 + (size_t)l * H_, tmp2,
                    MTOT, H_, FF_, FF_, H_, H_,
                    0, 0, 0, 0, 0, 0, 1, 1, 1.f, 0, stream);

        // hidden = LN2(hidden + ff)
        add_ln_kernel<<<MTOT, 256, 0, stream>>>(hidden, tmp2, ln2_g + (size_t)l * H_, ln2_b + (size_t)l * H_);
    }

    // ---- distance head: relu(hidden @ wd1 + bd1) @ wd2 + bd2 ----
    launch_gemm(0, hidden, wd1, bd1, tmp1, MTOT, H_, H_, H_, H_, H_,
                0, 0, 0, 0, 0, 0, 1, 1, 1.f, 1, stream);
    launch_gemm(0, tmp1, wd2, bd2, out_dist, MTOT, H_, H_, H_, H_, H_,
                0, 0, 0, 0, 0, 0, 1, 1, 1.f, 0, stream);

    // ---- angle head: relu(hidden @ wa1 + ba1) @ wa2 + ba2 ----
    launch_gemm(0, hidden, wa1, ba1, tmp1, MTOT, H_, H_, H_, H_, H_,
                0, 0, 0, 0, 0, 0, 1, 1, 1.f, 1, stream);
    {
        long long tot = (long long)MTOT * 3;
        small_gemm_kernel<<<(unsigned)((tot + 255) / 256), 256, 0, stream>>>(
            tmp1, wa2, ba2, out_ang, MTOT, H_, 3);
    }

    // ---- coordinate scan (sequential per batch element) ----
    coords_kernel<<<1, B_, 0, stream>>>(out_dist, out_ang, out_coord);

    // ---- final hidden to output ----
    hipMemcpyAsync(out_hidden, hidden, (size_t)MTOT * H_ * sizeof(float),
                   hipMemcpyDeviceToDevice, stream);
}
